// SparseAutoencoder_83365315215361
// MI455X (gfx1250) — matrix operand order repaired and checked by reading, not compiled, not run on hardware
//
#include <hip/hip_runtime.h>
#include <hip/hip_bf16.h>

// ---------------------------------------------------------------------------
// SAE forward for MI455X (gfx1250, wave32, WMMA).
//   pre   = x @ W_enc^T + b_enc        [8192 x 16384]   (bf16 WMMA, f32 accum)
//   h     = topk64(relu(pre))          (register-resident radix select)
//   recon = h @ W_dec^T                (sparse gather vs W_dec^T, L2-resident)
// d_out = [recon (8192*1024 f32) | h (8192*16384 f32)]
// ---------------------------------------------------------------------------

#define D_MODEL 1024
#define D_SAE   16384
#define NROWS   8192
#define TOPK    64

#define KTILE 32
#define MBLK  128   // rows of x per block (4 wave-rows * 32)
#define NBLK  128   // SAE features per block (2 wave-cols * 64)
#define NT    (D_MODEL / KTILE)   // 32 K-steps

#ifndef __has_builtin
#define __has_builtin(x) 0
#endif
#if __has_builtin(__builtin_amdgcn_global_load_async_to_lds_b64)
#define USE_ASYNC_LDS 1
#else
#define USE_ASYNC_LDS 0
#endif

typedef __attribute__((ext_vector_type(16))) __bf16 v16bf;
typedef __attribute__((ext_vector_type(8)))  float  v8f;
typedef int v2i_g __attribute__((vector_size(8)));   // matches builtin param pointee

// async copy: global (AS1) -> LDS (AS3), 8 bytes per lane, ASYNCcnt-tracked
#define ASYNC_CP_B64(gp, lp)                                                   \
  __builtin_amdgcn_global_load_async_to_lds_b64(                               \
      (__attribute__((address_space(1))) v2i_g*)(gp),                          \
      (__attribute__((address_space(3))) v2i_g*)(lp), 0, 0)

union BF16x16 { unsigned short u[16]; v16bf v; };

__device__ __forceinline__ void wait_async_zero() {
#if __has_builtin(__builtin_amdgcn_s_wait_asynccnt)
  __builtin_amdgcn_s_wait_asynccnt(0);
#else
  asm volatile("s_wait_asynccnt 0x0" ::: "memory");
#endif
}

// ---------------------------------------------------------------------------
// f32 -> bf16 (round to nearest even), grid-stride
// ---------------------------------------------------------------------------
__global__ void f32_to_bf16(const float* __restrict__ src,
                            unsigned short* __restrict__ dst, int nElems) {
  int i = blockIdx.x * blockDim.x + threadIdx.x;
  int stride = gridDim.x * blockDim.x;
  for (; i < nElems; i += stride) {
    unsigned u = __float_as_uint(src[i]);
    unsigned r = (u + 0x7FFFu + ((u >> 16) & 1u)) >> 16;
    dst[i] = (unsigned short)r;
  }
}

// ---------------------------------------------------------------------------
// W_dec [D_MODEL][D_SAE] -> W_decT [D_SAE][D_MODEL]  (tiled 32x32 via LDS)
// ---------------------------------------------------------------------------
__global__ __launch_bounds__(256) void transpose_wdec(
    const float* __restrict__ Wd, float* __restrict__ WdT) {
  __shared__ float tile[32][33];
  const int sBase = blockIdx.x * 32;
  const int dBase = blockIdx.y * 32;
  const int tx = threadIdx.x & 31;
  const int ty = threadIdx.x >> 5;  // 0..7
#pragma unroll
  for (int i = 0; i < 4; ++i) {
    int d = dBase + ty + i * 8;
    tile[ty + i * 8][tx] = Wd[(size_t)d * D_SAE + sBase + tx];
  }
  __syncthreads();
#pragma unroll
  for (int i = 0; i < 4; ++i) {
    int s = sBase + ty + i * 8;
    WdT[(size_t)s * D_MODEL + dBase + tx] = tile[tx][ty + i * 8];
  }
}

// ---------------------------------------------------------------------------
// Encoder GEMM: h_dense = relu(x @ W_enc^T + b_enc)
// 256 threads = 8 waves in a 4(M) x 2(N) grid. Block tile 128x128, K step 32.
// Each wave: 2 A-frags x 4 B-frags -> 8 WMMAs per K-step.
// Double-buffered LDS; async global->LDS copies when the builtin exists.
// ---------------------------------------------------------------------------
__global__ __launch_bounds__(256) void enc_gemm(
    const unsigned short* __restrict__ xb,   // [NROWS][D_MODEL] bf16 bits
    const unsigned short* __restrict__ wb,   // [D_SAE][D_MODEL] bf16 bits
    const float* __restrict__ b_enc,         // [D_SAE]
    float* __restrict__ hout)                // [NROWS][D_SAE]
{
  __shared__ unsigned short As[2][MBLK * KTILE];  // 2 x 8 KB
  __shared__ unsigned short Bs[2][NBLK * KTILE];  // 2 x 8 KB

  const int s0   = blockIdx.x * NBLK;
  const int m0   = blockIdx.y * MBLK;
  const int t    = threadIdx.x;
  const int wave = t >> 5;
  const int lane = t & 31;
  const int half = lane >> 4;   // EXEC all-1s: no divergence around WMMA
  const int l16  = lane & 15;
  const int wm   = wave & 3;    // 0..3 : 32-row strip
  const int wn   = wave >> 2;   // 0..1 : 64-col strip

  v8f acc[2][4];
#pragma unroll
  for (int f = 0; f < 2; ++f)
#pragma unroll
    for (int j = 0; j < 4; ++j)
      acc[f][j] = (v8f){0.f,0.f,0.f,0.f,0.f,0.f,0.f,0.f};

  // Tile copy plan: both tiles are 128 rows x 32 bf16 = 1024 u64 chunks;
  // 4 chunks per thread, coalesced.
  int rC[4], cC[4];
#pragma unroll
  for (int p = 0; p < 4; ++p) {
    int c = t + p * 256;
    rC[p] = c >> 3;            // row 0..127
    cC[p] = (c & 7) * 4;       // bf16 col offset (u64-granular)
  }

#if USE_ASYNC_LDS
  // ---- prologue: async copy tile 0 into buffer 0 ----
#pragma unroll
  for (int p = 0; p < 4; ++p)
    ASYNC_CP_B64(xb + (size_t)(m0 + rC[p]) * D_MODEL + cC[p],
                 &As[0][rC[p] * KTILE + cC[p]]);
#pragma unroll
  for (int p = 0; p < 4; ++p)
    ASYNC_CP_B64(wb + (size_t)(s0 + rC[p]) * D_MODEL + cC[p],
                 &Bs[0][rC[p] * KTILE + cC[p]]);
  wait_async_zero();
#else
  {
    unsigned long long ra[4], rb[4];
#pragma unroll
    for (int p = 0; p < 4; ++p) {
      ra[p] = *(const unsigned long long*)(xb + (size_t)(m0 + rC[p]) * D_MODEL + cC[p]);
      rb[p] = *(const unsigned long long*)(wb + (size_t)(s0 + rC[p]) * D_MODEL + cC[p]);
    }
#pragma unroll
    for (int p = 0; p < 4; ++p) {
      *(unsigned long long*)&As[0][rC[p] * KTILE + cC[p]] = ra[p];
      *(unsigned long long*)&Bs[0][rC[p] * KTILE + cC[p]] = rb[p];
    }
  }
#endif
  __syncthreads();

  for (int kt = 0; kt < NT; ++kt) {
    const int cur = kt & 1;
    const int nk0 = (kt + 1) * KTILE;

    // ---- prefetch next tile into the other buffer ----
#if USE_ASYNC_LDS
    if (kt + 1 < NT) {
#pragma unroll
      for (int p = 0; p < 4; ++p)
        ASYNC_CP_B64(xb + (size_t)(m0 + rC[p]) * D_MODEL + nk0 + cC[p],
                     &As[cur ^ 1][rC[p] * KTILE + cC[p]]);
#pragma unroll
      for (int p = 0; p < 4; ++p)
        ASYNC_CP_B64(wb + (size_t)(s0 + rC[p]) * D_MODEL + nk0 + cC[p],
                     &Bs[cur ^ 1][rC[p] * KTILE + cC[p]]);
    }
#else
    unsigned long long ra[4], rb[4];
    if (kt + 1 < NT) {
#pragma unroll
      for (int p = 0; p < 4; ++p) {
        ra[p] = *(const unsigned long long*)(xb + (size_t)(m0 + rC[p]) * D_MODEL + nk0 + cC[p]);
        rb[p] = *(const unsigned long long*)(wb + (size_t)(s0 + rC[p]) * D_MODEL + nk0 + cC[p]);
      }
    }
#endif

    // ---- compute from current buffer: 8 WMMAs ----
    // A frag (16x32 MxK): element e -> K = (e/8)*16 + 8*half + (e%8)
    BF16x16 a[2];
#pragma unroll
    for (int f = 0; f < 2; ++f)
#pragma unroll
      for (int e = 0; e < 16; ++e) {
        int kk = ((e >> 3) << 4) + (half << 3) + (e & 7);
        a[f].u[e] = As[cur][(wm * 32 + f * 16 + l16) * KTILE + kk];
      }
#pragma unroll
    for (int j = 0; j < 4; ++j) {
      // B frag (32x16 KxN): column n=l16, element e -> K = (e/8)*16 + 8*half + (e%8)
      BF16x16 b;
#pragma unroll
      for (int e = 0; e < 16; ++e)
        b.u[e] = Bs[cur][(wn * 64 + j * 16 + l16) * KTILE + ((e >> 3) << 4) + (half << 3) + (e & 7)];
#pragma unroll
      for (int f = 0; f < 2; ++f)
        acc[f][j] = __builtin_amdgcn_wmma_f32_16x16x32_bf16(
            false, a[f].v, false, b.v, (short)0, acc[f][j], false, false);
    }

    // ---- publish next buffer ----
#if USE_ASYNC_LDS
    if (kt + 1 < NT) wait_async_zero();
#else
    if (kt + 1 < NT) {
#pragma unroll
      for (int p = 0; p < 4; ++p) {
        *(unsigned long long*)&As[cur ^ 1][rC[p] * KTILE + cC[p]] = ra[p];
        *(unsigned long long*)&Bs[cur ^ 1][rC[p] * KTILE + cC[p]] = rb[p];
      }
    }
#endif
    __syncthreads();
  }

  // Epilogue: bias + ReLU; C/D layout: n = l16, m = v + 8*half
#pragma unroll
  for (int f = 0; f < 2; ++f)
#pragma unroll
    for (int j = 0; j < 4; ++j) {
      int s = s0 + wn * 64 + j * 16 + l16;
      float bias = b_enc[s];
#pragma unroll
      for (int v = 0; v < 8; ++v) {
        int m = wm * 32 + f * 16 + v + 8 * half;
        float val = acc[f][j][v] + bias;
        val = val > 0.f ? val : 0.f;
        hout[(size_t)(m0 + m) * D_SAE + s] = val;
      }
    }
}

// ---------------------------------------------------------------------------
// Per-row top-64 + sparse decode. One block (256 thr) per row; the 16384-value
// row lives in REGISTERS (64/thread): radix-select passes are pure VALU.
// Post-ReLU values >= 0 so uint compare == float compare.
// ---------------------------------------------------------------------------
__global__ __launch_bounds__(256) void topk_decode(
    float* __restrict__ h,           // [NROWS][D_SAE] in: dense relu(pre); out: masked
    const float* __restrict__ WdT,   // [D_SAE][D_MODEL]
    float* __restrict__ recon)       // [NROWS][D_MODEL]
{
  __shared__ int   s_cnt;
  __shared__ int   s_cGt, s_cEq;
  __shared__ int   actIdx[TOPK];
  __shared__ float actVal[TOPK];

  const int n = blockIdx.x;
  const int t = threadIdx.x;
  float* hrow = h + (size_t)n * D_SAE;

  float vals[64];
#pragma unroll
  for (int j = 0; j < 64; ++j) vals[j] = hrow[t + j * 256];

  // Greedy MSB-first search for max thr with count(bits >= thr) >= TOPK.
  unsigned thr = 0;
  for (int bit = 30; bit >= 0; --bit) {
    unsigned cand = thr | (1u << bit);
    if (t == 0) s_cnt = 0;
    __syncthreads();
    int local = 0;
#pragma unroll
    for (int j = 0; j < 64; ++j)
      local += (__float_as_uint(vals[j]) >= cand) ? 1 : 0;
    atomicAdd(&s_cnt, local);
    __syncthreads();
    if (s_cnt >= TOPK) thr = cand;   // uniform across block
    __syncthreads();
  }

  // g = count strictly greater (guaranteed < TOPK); budget TOPK-g for ties
  if (t == 0) s_cnt = 0;
  __syncthreads();
  int localg = 0;
#pragma unroll
  for (int j = 0; j < 64; ++j)
    localg += (__float_as_uint(vals[j]) > thr) ? 1 : 0;
  atomicAdd(&s_cnt, localg);
  __syncthreads();
  const int g = s_cnt;
  __syncthreads();

  if (t == 0) { s_cGt = 0; s_cEq = 0; }
  __syncthreads();
  unsigned long long keepMask = 0ull;
#pragma unroll
  for (int j = 0; j < 64; ++j) {
    unsigned b = __float_as_uint(vals[j]);
    int i = t + j * 256;
    if (b > thr) {
      int p = atomicAdd(&s_cGt, 1);
      actIdx[p] = i; actVal[p] = vals[j];
      keepMask |= (1ull << j);
    } else if (b == thr) {
      int q = atomicAdd(&s_cEq, 1);
      if (q < TOPK - g) {
        actIdx[g + q] = i; actVal[g + q] = vals[j];
        keepMask |= (1ull << j);
      }
    }
  }
  __syncthreads();

  // Masked write-back straight to global (coalesced)
#pragma unroll
  for (int j = 0; j < 64; ++j)
    hrow[t + j * 256] = ((keepMask >> j) & 1ull) ? vals[j] : 0.f;

  // Decode: recon[n,d] = sum_j actVal[j] * WdT[actIdx[j]][d]; coalesced in d
  float accd[4] = {0.f, 0.f, 0.f, 0.f};
  for (int j = 0; j < TOPK; ++j) {
    int s = actIdx[j];
    float v = actVal[j];
    const float* wcol = WdT + (size_t)s * D_MODEL;
#pragma unroll
    for (int d = 0; d < 4; ++d)
      accd[d] += v * wcol[t + d * 256];
  }
  float* rrow = recon + (size_t)n * D_MODEL;
#pragma unroll
  for (int d = 0; d < 4; ++d) rrow[t + d * 256] = accd[d];
}

// ---------------------------------------------------------------------------
// Launch
// ---------------------------------------------------------------------------
extern "C" void kernel_launch(void* const* d_in, const int* in_sizes, int n_in,
                              void* d_out, int out_size, void* d_ws, size_t ws_size,
                              hipStream_t stream) {
  const float* x     = (const float*)d_in[0];  // [NROWS][D_MODEL]
  const float* W_enc = (const float*)d_in[1];  // [D_SAE][D_MODEL]
  const float* b_enc = (const float*)d_in[2];  // [D_SAE]
  const float* W_dec = (const float*)d_in[3];  // [D_MODEL][D_SAE]

  float* recon = (float*)d_out;                              // [NROWS][D_MODEL]
  float* h     = (float*)d_out + (size_t)NROWS * D_MODEL;    // [NROWS][D_SAE]

  // Workspace: xb (16MB bf16) | wb (32MB bf16) | WdT (64MB f32)
  unsigned short* xb = (unsigned short*)d_ws;
  unsigned short* wb = xb + (size_t)NROWS * D_MODEL;
  float* WdT = (float*)(wb + (size_t)D_SAE * D_MODEL);

  f32_to_bf16<<<2048, 256, 0, stream>>>(x, xb, NROWS * D_MODEL);
  f32_to_bf16<<<4096, 256, 0, stream>>>(W_enc, wb, D_SAE * D_MODEL);
  transpose_wdec<<<dim3(D_SAE / 32, D_MODEL / 32), 256, 0, stream>>>(W_dec, WdT);

  enc_gemm<<<dim3(D_SAE / NBLK, NROWS / MBLK), 256, 0, stream>>>(xb, wb, b_enc, h);

  topk_decode<<<NROWS, 256, 0, stream>>>(h, WdT, recon);
}
